// ComplexDeepDepthwiseCNN_BSS_26182120636746
// MI455X (gfx1250) — matrix operand order repaired and checked by reading, not compiled, not run on hardware
//
#include <hip/hip_runtime.h>

// ---------------- problem constants ----------------
#define FBINS   257
#define NCHN    8
#define NSRC    2
#define KW      5
#define HID     32
#define TLEN    2000
#define BATCH   4

// ---------------- tiling ----------------
#define TT        512                 // timesteps per workgroup
#define HALO      8                   // 3 conv layers x pad 2, + slack
#define COLS      (TT + 2 * HALO)     // 528
#define NTILES    (COLS / 16)         // 33
#define NTHREADS  256
#define NWAVES    (NTHREADS / 32)

// column-major activation strides (elements), padded for conflict-free LDS banks
#define ZSTR   24   // 16 rows (zr0-7, zi0-7) padded -> 48 B/col  (12 dwords)
#define HSTR   72   // 64 rows padded            -> 144 B/col (36 dwords)
#define H3STR  36   // 32 f32 rows padded        -> 144 B/col (36 dwords)

typedef __attribute__((ext_vector_type(16))) __bf16 v16bf;
typedef __attribute__((ext_vector_type(8)))  float  v8f;

union FragA { float4 f4[2]; v16bf v; };
union FragB { float4 f4[2]; v16bf v; };
union PackD { __bf16 h[8]; float4 f4; };
union SplitD { v8f v; float4 f4[2]; };

__device__ __forceinline__ unsigned short f2bf(float f) {
  unsigned int u = __float_as_uint(f);
  u += 0x7FFFu + ((u >> 16) & 1u);   // round-to-nearest-even
  return (unsigned short)(u >> 16);
}
__device__ __forceinline__ float bf2f(unsigned short h) {
  return __uint_as_float(((unsigned int)h) << 16);
}

// ---------------- LDS layout (bytes) ----------------
#define Z_OFF    0                                   // ushort [COLS][ZSTR]
#define A1_OFF   (Z_OFF  + COLS * ZSTR * 2)          // ushort [64][96]   row-major
#define A2_OFF   (A1_OFF + 64 * 96  * 2)             // ushort [64][320]  row-major
#define A3_OFF   (A2_OFF + 64 * 320 * 2)             // ushort [32][320]  row-major
#define H1_OFF   (A3_OFF + 32 * 320 * 2)             // ushort [COLS][HSTR]
#define H2_OFF   (H1_OFF + COLS * HSTR * 2)          // ushort [COLS][HSTR]
#define H3_OFF   (H2_OFF + COLS * HSTR * 2)          // float  [COLS][H3STR]
#define SMEM_BYTES (H3_OFF + COLS * H3STR * 4)       // = 327168 <= 320 KB

static_assert(SMEM_BYTES <= 320 * 1024, "LDS budget exceeded");
static_assert((A1_OFF % 16) == 0 && (A2_OFF % 16) == 0 && (A3_OFF % 16) == 0 &&
              (H1_OFF % 16) == 0 && (H2_OFF % 16) == 0 && (H3_OFF % 16) == 0,
              "16B alignment required for ds_load_b128/ds_store_b128");

// One conv layer as GEMM: D = A(MTILES*16 x KPAD) x im2col(src), tap-major K.
//   K index: k = w * R + pc   (R = channel rows per tap)
//   src/dst column-major [col][row]: B fragments and D tiles contiguous per lane.
// A wave owns NBLK N-tiles x all M-tiles: per K-tile, A fragments (MTILES*2 b128)
// are loaded once and feed MTILES*NBLK WMMAs; B = NBLK*2 b128.
template<int KPAD, int R, int SSTR, int DSTR, int MTILES, int KTILES, int NBLK, bool F32OUT>
__device__ __forceinline__ void gemm_layer(const unsigned short* __restrict__ A,
                                           const unsigned short* __restrict__ src,
                                           void* __restrict__ dstv,
                                           int wave, int lane, int t0)
{
  const int half = (lane >> 4) & 1;
  const int lm   = lane & 15;

  for (int nt0 = wave * NBLK; nt0 < NTILES; nt0 += NWAVES * NBLK) {
    int ncol[NBLK];
    #pragma unroll
    for (int nb = 0; nb < NBLK; ++nb) {
      int nt = nt0 + nb;
      if (nt > NTILES - 1) nt = NTILES - 1;   // overflow block: duplicate last tile
      ncol[nb] = nt * 16 + lm;
    }

    v8f acc[NBLK][MTILES];
    #pragma unroll
    for (int nb = 0; nb < NBLK; ++nb)
      #pragma unroll
      for (int mt = 0; mt < MTILES; ++mt) acc[nb][mt] = {};

    for (int kt = 0; kt < KTILES; ++kt) {
      // ---- A fragments (N-invariant): load once per K-tile ----
      FragA fa[MTILES];
      const int kh = kt * 32 + half * 16;
      #pragma unroll
      for (int mt = 0; mt < MTILES; ++mt) {
        const int m = mt * 16 + lm;
        const float4* ap = reinterpret_cast<const float4*>(A + (size_t)m * KPAD + kh);
        fa[mt].f4[0] = ap[0];      // K = kh .. kh+7
        fa[mt].f4[1] = ap[1];      // K = kh+8 .. kh+15
      }
      // ---- B fragment per N-block: 16 contiguous channel rows at one column ----
      const int k0  = kt * 32 + half * 16;
      const int w   = k0 / R;           // tap (compile-time R)
      const int pc0 = k0 - w * R;       // first channel row
      #pragma unroll
      for (int nb = 0; nb < NBLK; ++nb) {
        int sc = ncol[nb] + w - 2;      // PAD = 2
        sc = (sc < 0) ? 0 : ((sc > COLS - 1) ? COLS - 1 : sc);
        FragB fb;
        const float4* bp = reinterpret_cast<const float4*>(src + sc * SSTR + pc0);
        fb.f4[0] = bp[0];
        fb.f4[1] = bp[1];
        // (w out of range only where A's K-padding is zero -> product is zero)
        #pragma unroll
        for (int mt = 0; mt < MTILES; ++mt)
          acc[nb][mt] = __builtin_amdgcn_wmma_f32_16x16x32_bf16(false, fa[mt].v, false, fb.v,
                                                               (short)0, acc[nb][mt], false, false);
      }
    }

    // ---- D store: rows contiguous per lane in column-major dst ----
    #pragma unroll
    for (int nb = 0; nb < NBLK; ++nb) {
      const int tg = t0 - HALO + ncol[nb];
      const float msk = (tg >= 0 && tg < TLEN) ? 1.0f : 0.0f;  // inter-layer zero pad
      #pragma unroll
      for (int mt = 0; mt < MTILES; ++mt) {
        const int mrow = mt * 16 + half * 8;
        if (F32OUT) {
          float* dst = (float*)dstv;
          SplitD od; od.v = acc[nb][mt];
          float4* op = reinterpret_cast<float4*>(dst + ncol[nb] * DSTR + mrow);
          op[0] = od.f4[0];
          op[1] = od.f4[1];
        } else {
          unsigned short* dst = (unsigned short*)dstv;
          PackD od;
          #pragma unroll
          for (int v = 0; v < 8; ++v) {
            // ReLU via single v_med3, zero-mask via mul, native bf16 cvt
            const float r = __builtin_amdgcn_fmed3f(acc[nb][mt][v], 0.0f, __builtin_inff());
            od.h[v] = (__bf16)(r * msk);
          }
          *reinterpret_cast<float4*>(dst + ncol[nb] * DSTR + mrow) = od.f4;
        }
      }
    }
  }
}

extern "C" __global__ __launch_bounds__(NTHREADS)
void bss_complex_dwconv_kernel(const float* __restrict__ x,
                               const float* __restrict__ Wr1, const float* __restrict__ Wi1,
                               const float* __restrict__ Wr2, const float* __restrict__ Wi2,
                               const float* __restrict__ Wr3, const float* __restrict__ Wi3,
                               float* __restrict__ out)
{
  extern __shared__ __align__(16) unsigned char smem[];
  unsigned short* Zs = (unsigned short*)(smem + Z_OFF);
  unsigned short* A1 = (unsigned short*)(smem + A1_OFF);
  unsigned short* A2 = (unsigned short*)(smem + A2_OFF);
  unsigned short* A3 = (unsigned short*)(smem + A3_OFF);
  unsigned short* H1 = (unsigned short*)(smem + H1_OFF);
  unsigned short* H2 = (unsigned short*)(smem + H2_OFF);
  float*          H3 = (float*)(smem + H3_OFF);

  const int tid  = threadIdx.x;
  const int lane = tid & 31;
  const int wave = tid >> 5;
  const int f    = blockIdx.y;
  const int b    = blockIdx.z;
  const int t0   = blockIdx.x * TT;

  // ---- Phase 0a: stage x tile, column-major bf16, zero-padded, coalesced in t ----
  for (int i = tid; i < 16 * COLS; i += NTHREADS) {
    const int r   = i / COLS;          // 0..7 = zr ch, 8..15 = zi ch
    const int col = i - r * COLS;
    const int part = r >> 3;
    const int c    = r & 7;
    const int t    = t0 - HALO + col;
    float v = 0.0f;
    if (t >= 0 && t < TLEN)
      v = x[(((size_t)b * (2 * FBINS) + part * FBINS + f) * NCHN + c) * TLEN + t];
    Zs[col * ZSTR + r] = f2bf(v);
  }
  // ---- Phase 0b: pack block-complex A matrices, tap-major K (k = w*R + pc) ----
  // A1: 64 x 96, R = 16 (8 zr + 8 zi), K 80..95 zero pad
  for (int i = tid; i < 64 * 96; i += NTHREADS) {
    const int m = i / 96, k = i - m * 96;
    const int w = k / 16, pc = k - w * 16;
    float v = 0.0f;
    if (w < KW) {
      const int co = (m < HID) ? m : m - HID;
      const int ci = (pc < NCHN) ? pc : pc - NCHN;
      const size_t off = (((size_t)f * HID + co) * NCHN + ci) * KW + w;
      const float wr = Wr1[off], wi = Wi1[off];
      v = (m < HID) ? ((pc < NCHN) ? wr : -wi) : ((pc < NCHN) ? wi : wr);
    }
    A1[i] = f2bf(v);
  }
  // A2: 64 x 320, R = 64 (32 h1r + 32 h1i)
  for (int i = tid; i < 64 * 320; i += NTHREADS) {
    const int m = i / 320, k = i - m * 320;
    const int w = k / 64, pc = k - w * 64;
    const int co = (m < HID) ? m : m - HID;
    const int ci = (pc < HID) ? pc : pc - HID;
    const size_t off = (((size_t)f * HID + co) * HID + ci) * KW + w;
    const float wr = Wr2[off], wi = Wi2[off];
    A2[i] = f2bf((m < HID) ? ((pc < HID) ? wr : -wi) : ((pc < HID) ? wi : wr));
  }
  // A3: 32 x 320 (Cout = NCHN*NSRC = 16 per part), R = 64
  for (int i = tid; i < 32 * 320; i += NTHREADS) {
    const int m = i / 320, k = i - m * 320;
    const int w = k / 64, pc = k - w * 64;
    const int co = (m < 16) ? m : m - 16;
    const int ci = (pc < HID) ? pc : pc - HID;
    const size_t off = (((size_t)f * (NCHN * NSRC) + co) * HID + ci) * KW + w;
    const float wr = Wr3[off], wi = Wi3[off];
    A3[i] = f2bf((m < 16) ? ((pc < HID) ? wr : -wi) : ((pc < HID) ? wi : wr));
  }
  __syncthreads();

  // ---- Layer 1: (64 x 96) x (96 x COLS) -> relu -> H1 ----
  gemm_layer<96, 16, ZSTR, HSTR, 4, 3, 2, false>(A1, Zs, H1, wave, lane, t0);
  __syncthreads();
  // ---- Layer 2: (64 x 320) x (320 x COLS) -> relu -> H2 ----
  gemm_layer<320, 64, HSTR, HSTR, 4, 10, 2, false>(A2, H1, H2, wave, lane, t0);
  __syncthreads();
  // ---- Layer 3: (32 x 320) x (320 x COLS) -> H3 (f32, no relu) ----
  gemm_layer<320, 64, HSTR, H3STR, 2, 10, 4, true>(A3, H2, H3, wave, lane, t0);
  __syncthreads();

  // ---- Mask apply: Y = sum_c x * h (complex), reduce over channels ----
  for (int i = tid; i < NSRC * TT; i += NTHREADS) {
    const int s  = i / TT;
    const int tt = i - s * TT;
    const int t  = t0 + tt;
    if (t >= TLEN) continue;
    const int col = tt + HALO;
    float yr = 0.0f, yi = 0.0f;
    #pragma unroll
    for (int c = 0; c < NCHN; ++c) {
      const float xr = bf2f(Zs[col * ZSTR + c]);
      const float xi = bf2f(Zs[col * ZSTR + NCHN + c]);
      const float hr = H3[col * H3STR + (c * NSRC + s)];        // rows 0..15  = real
      const float hi = H3[col * H3STR + 16 + (c * NSRC + s)];   // rows 16..31 = imag
      yr += xr * hr - xi * hi;
      yi += xr * hi + xi * hr;
    }
    out[(((size_t)b * (2 * FBINS) + f) * NSRC + s) * TLEN + t]         = yr;
    out[(((size_t)b * (2 * FBINS) + FBINS + f) * NSRC + s) * TLEN + t] = yi;
  }
}

extern "C" void kernel_launch(void* const* d_in, const int* in_sizes, int n_in,
                              void* d_out, int out_size, void* d_ws, size_t ws_size,
                              hipStream_t stream) {
  const float* x   = (const float*)d_in[0];
  const float* Wr1 = (const float*)d_in[1];
  const float* Wi1 = (const float*)d_in[2];
  const float* Wr2 = (const float*)d_in[3];
  const float* Wi2 = (const float*)d_in[4];
  const float* Wr3 = (const float*)d_in[5];
  const float* Wi3 = (const float*)d_in[6];
  float* out = (float*)d_out;

  (void)in_sizes; (void)n_in; (void)out_size; (void)d_ws; (void)ws_size;

  (void)hipFuncSetAttribute((const void*)bss_complex_dwconv_kernel,
                            hipFuncAttributeMaxDynamicSharedMemorySize, SMEM_BYTES);

  dim3 grid((TLEN + TT - 1) / TT, FBINS, BATCH);  // 4 x 257 x 4 = 4112 workgroups
  bss_complex_dwconv_kernel<<<grid, NTHREADS, SMEM_BYTES, stream>>>(
      x, Wr1, Wi1, Wr2, Wi2, Wr3, Wi3, out);
}
